// Decoder_52201032516290
// MI455X (gfx1250) — matrix operand order repaired and checked by reading, not compiled, not run on hardware
//
#include <hip/hip_runtime.h>
#include <hip/hip_bf16.h>

// ---------------------------------------------------------------------------
// Edge MLP decoder:  out[e] = W2 . relu(W1 @ concat(z[row[e]], z[col[e]]) + b1) + b2
// GEMM1: M=E (3.2M), N=128, K=128  -> v_wmma_f32_16x16x32_f16, f32 accum.
//   - z pre-converted to f16 in workspace (L2-resident, halves gather bytes)
//   - W1 pre-swizzled into tile-major WMMA B layout -> conflict-free ds_load_b128
//   - W1 staged to LDS via GLOBAL_LOAD_ASYNC_TO_LDS_B128 when available
// ---------------------------------------------------------------------------

typedef __attribute__((ext_vector_type(16))) _Float16 v16h;
typedef __attribute__((ext_vector_type(8)))  _Float16 v8h;
typedef __attribute__((ext_vector_type(4)))  _Float16 v4h;
typedef __attribute__((ext_vector_type(8)))  float    v8f;
typedef __attribute__((ext_vector_type(4)))  float    v4f;

// type expected by the async-LDS builtin (per hipcc diagnostic):
typedef int v4i_vs __attribute__((vector_size(16)));
typedef __attribute__((address_space(1))) v4i_vs* gas1_p;
typedef __attribute__((address_space(3))) v4i_vs* las3_p;

#define ZDIM 64
#define NHID 128

#if defined(__HIP_DEVICE_COMPILE__) &&                                   \
    __has_builtin(__builtin_amdgcn_global_load_async_to_lds_b128) &&     \
    __has_builtin(__builtin_amdgcn_s_wait_asynccnt)
#define EDGE_ASYNC_LDS 1
#else
#define EDGE_ASYNC_LDS 0
#endif

// --- W1 f32 -> f16, swizzled into tile-major WMMA B layout -----------------
// Tile (kb,nb) holds B[32x16] for K block kb, N block nb: 512 halves = 1KB.
// Within a tile, lane L owns 16 consecutive halves at L*16:
//   L<16 : N = nb*16+L,      K = kb*32 + 0..7, 16..23
//   L>=16: N = nb*16+(L-16), K = kb*32 + 8..15, 24..31
__global__ void w1_swizzle(const float* __restrict__ w1,
                           _Float16* __restrict__ w1s) {
  int idx = blockIdx.x * 256 + threadIdx.x;   // 0..16383, = n*128 + k
  int n = idx >> 7;
  int k = idx & 127;
  int nb = n >> 4, mloc = n & 15;
  int kb = k >> 5, kr = k & 31;
  int lane = mloc + ((kr & 8) ? 16 : 0);
  int i = (kr & 7) | ((kr >> 1) & 8);
  w1s[((kb * 8 + nb) << 9) + (lane << 4) + i] = (_Float16)w1[idx];
}

// --- z f32 -> f16 (6.4M elems, 4 per thread) -------------------------------
__global__ void z_to_f16(const float* __restrict__ z,
                         _Float16* __restrict__ zh, int n4) {
  int i = blockIdx.x * 256 + threadIdx.x;
  if (i < n4) {
    v4f v = ((const v4f*)z)[i];
    v4h o;
#pragma unroll
    for (int j = 0; j < 4; ++j) o[j] = (_Float16)v[j];
    ((v4h*)zh)[i] = o;
  }
}

// A-tile slice from an f32 z row (fallback path), per ISA 16-bit A layout:
//   elements 0..7 = K k0..k0+7,  8..15 = K k0+16..k0+23  (k0 pre-shifted +8 hi)
__device__ __forceinline__ v16h load_a_f32(const float* __restrict__ src, int k0) {
  v4f q0 = *(const v4f*)(src + k0);
  v4f q1 = *(const v4f*)(src + k0 + 4);
  v4f q2 = *(const v4f*)(src + k0 + 16);
  v4f q3 = *(const v4f*)(src + k0 + 20);
  v16h a;
#pragma unroll
  for (int i = 0; i < 4; ++i) {
    a[i]      = (_Float16)q0[i];
    a[4 + i]  = (_Float16)q1[i];
    a[8 + i]  = (_Float16)q2[i];
    a[12 + i] = (_Float16)q3[i];
  }
  return a;
}

// A-tile slice from an f16 z row: two 16B loads, no conversion
__device__ __forceinline__ v16h load_a_f16(const _Float16* __restrict__ src, int k0) {
  v8h lo = *(const v8h*)(src + k0);
  v8h hi = *(const v8h*)(src + k0 + 16);
  v16h a;
#pragma unroll
  for (int i = 0; i < 8; ++i) { a[i] = lo[i]; a[8 + i] = hi[i]; }
  return a;
}

template <bool ZF16>
__launch_bounds__(256, 1)
__global__ void edge_mlp(const float* __restrict__ zf,
                         const _Float16* __restrict__ zh,
                         const int* __restrict__ row,
                         const int* __restrict__ col,
                         const _Float16* __restrict__ w1s,  // tile-major f16
                         const float* __restrict__ b1,
                         const float* __restrict__ w2,
                         const float* __restrict__ b2,
                         float* __restrict__ out,
                         int E) {
  __shared__ __align__(32) _Float16 sW1[NHID * NHID];  // 32 KB, tile-major
  __shared__ float sB1[NHID];
  __shared__ float sW2[NHID];

  // ---- stage W1 (32KB) into LDS -------------------------------------------
#if EDGE_ASYNC_LDS
  {
    _Float16* w1nc = const_cast<_Float16*>(w1s);
    gas1_p gsrc = (gas1_p)w1nc;          // global src, 16B units
    las3_p ldst = (las3_p)sW1;           // LDS dst,   16B units
    const int t = threadIdx.x;           // 256 threads x 8 x 16B = 32KB
#pragma unroll
    for (int i = 0; i < 8; ++i)
      __builtin_amdgcn_global_load_async_to_lds_b128(
          gsrc + t + i * 256, ldst + t + i * 256, 0, 0);
    __builtin_amdgcn_s_wait_asynccnt(0);
  }
#else
  {
    const uint4* src = (const uint4*)w1s;
    uint4* dst = (uint4*)sW1;
#pragma unroll
    for (int i = 0; i < 8; ++i)
      dst[threadIdx.x + i * 256] = src[threadIdx.x + i * 256];
  }
#endif
  if (threadIdx.x < NHID) sB1[threadIdx.x] = b1[threadIdx.x];
  else                    sW2[threadIdx.x - NHID] = w2[threadIdx.x - NHID];
  __syncthreads();

  const int lane   = threadIdx.x & 31;
  const int wave   = threadIdx.x >> 5;
  const int mloc   = lane & 15;
  const bool hi    = lane >= 16;
  const int kshift = hi ? 8 : 0;   // A-layout lane-half K shift

  const long wbase = (long)blockIdx.x * 256 + (long)wave * 32;  // 32 edges/wave

  long e0l = wbase + mloc;       if (e0l >= E) e0l = E - 1;
  long e1l = wbase + 16 + mloc;  if (e1l >= E) e1l = E - 1;
  const long ir0 = (long)row[e0l] * ZDIM;
  const long ic0 = (long)col[e0l] * ZDIM;
  const long ir1 = (long)row[e1l] * ZDIM;
  const long ic1 = (long)col[e1l] * ZDIM;

  // per-lane B base: lane's 32B slice inside each 1KB tile
  const _Float16* bbase = sW1 + (lane << 4);

  v8f zero = {};
  v8f acc[2][8];
#pragma unroll
  for (int mh = 0; mh < 2; ++mh)
#pragma unroll
    for (int nb = 0; nb < 8; ++nb) acc[mh][nb] = zero;

  // K = 128 in 4 steps of 32; kb 0,1 -> z[row], kb 2,3 -> z[col]
#pragma unroll
  for (int kb = 0; kb < 4; ++kb) {
    const long s0 = (kb < 2) ? ir0 : ic0;
    const long s1 = (kb < 2) ? ir1 : ic1;
    const int koff = (kb & 1) * 32 + kshift;
    v16h a0, a1;
    if (ZF16) {
      a0 = load_a_f16(zh + s0, koff);
      a1 = load_a_f16(zh + s1, koff);
    } else {
      a0 = load_a_f32(zf + s0, koff);
      a1 = load_a_f32(zf + s1, koff);
    }
#pragma unroll
    for (int nb = 0; nb < 8; ++nb) {
      v16h bm = *(const v16h*)(bbase + ((kb * 8 + nb) << 9));
      acc[0][nb] = __builtin_amdgcn_wmma_f32_16x16x32_f16(
          false, a0, false, bm, (short)0, acc[0][nb], false, false);
      acc[1][nb] = __builtin_amdgcn_wmma_f32_16x16x32_f16(
          false, a1, false, bm, (short)0, acc[1][nb], false, false);
    }
  }

  // fused bias + relu + (h . W2): C element j is (M = j or j+8, N = nb*16+mloc)
  float red[2][8];
#pragma unroll
  for (int mh = 0; mh < 2; ++mh)
#pragma unroll
    for (int j = 0; j < 8; ++j) red[mh][j] = 0.f;

#pragma unroll
  for (int nb = 0; nb < 8; ++nb) {
    const int ng = nb * 16 + mloc;
    const float b1v = sB1[ng];
    const float w2v = sW2[ng];
#pragma unroll
    for (int mh = 0; mh < 2; ++mh)
#pragma unroll
      for (int j = 0; j < 8; ++j) {
        float h = acc[mh][nb][j] + b1v;
        h = h > 0.f ? h : 0.f;
        red[mh][j] += h * w2v;
      }
  }

  // reduce over N across each 16-lane half (xor off<16 stays inside the half)
#pragma unroll
  for (int off = 1; off < 16; off <<= 1)
#pragma unroll
    for (int mh = 0; mh < 2; ++mh)
#pragma unroll
      for (int j = 0; j < 8; ++j)
        red[mh][j] += __shfl_xor(red[mh][j], off, 32);

  const float b2v = b2[0];
  if (mloc == 0) {                 // lanes 0 and 16 write 16 edges each
    const int mbase = hi ? 8 : 0;  // lane 0 -> M=j, lane 16 -> M=8+j
#pragma unroll
    for (int mh = 0; mh < 2; ++mh) {
      long ebase = wbase + mh * 16 + mbase;
      if (ebase < E) {
        float4 lo = make_float4(red[mh][0] + b2v, red[mh][1] + b2v,
                                red[mh][2] + b2v, red[mh][3] + b2v);
        float4 hx = make_float4(red[mh][4] + b2v, red[mh][5] + b2v,
                                red[mh][6] + b2v, red[mh][7] + b2v);
        *(float4*)(out + ebase)     = lo;
        *(float4*)(out + ebase + 4) = hx;
      }
    }
  }
}

extern "C" void kernel_launch(void* const* d_in, const int* in_sizes, int n_in,
                              void* d_out, int out_size, void* d_ws, size_t ws_size,
                              hipStream_t stream) {
  (void)n_in; (void)out_size;
  const float* z  = (const float*)d_in[0];
  const int*   rw = (const int*)d_in[1];
  const int*   cl = (const int*)d_in[2];
  const float* W1 = (const float*)d_in[3];
  const float* b1 = (const float*)d_in[4];
  const float* W2 = (const float*)d_in[5];
  const float* b2 = (const float*)d_in[6];
  float* out = (float*)d_out;

  const int E = in_sizes[1];

  const size_t w1Bytes = (size_t)NHID * NHID * sizeof(_Float16);  // 32 KB
  const size_t zElems  = (size_t)in_sizes[0];                     // nodes*64
  const size_t zBytes  = zElems * sizeof(_Float16);               // 12.8 MB

  _Float16* w1s = (_Float16*)d_ws;
  _Float16* zh  = (_Float16*)((char*)d_ws + w1Bytes);
  const bool useZ16 = ws_size >= w1Bytes + zBytes;

  w1_swizzle<<<(NHID * NHID) / 256, 256, 0, stream>>>(W1, w1s);

  const int blocks = (E + 255) / 256;  // 256 edges per block (8 waves x 32)
  if (useZ16) {
    const int n4 = (int)(zElems / 4);
    z_to_f16<<<(n4 + 255) / 256, 256, 0, stream>>>(z, zh, n4);
    edge_mlp<true><<<blocks, 256, 0, stream>>>(z, zh, rw, cl, w1s, b1, W2, b2,
                                               out, E);
  } else {
    edge_mlp<false><<<blocks, 256, 0, stream>>>(z, zh, rw, cl, w1s, b1, W2, b2,
                                                out, E);
  }
}
